// GCNN_5961414607260
// MI455X (gfx1250) — compile-verified, matrix operand order checked by reading, not run on hardware
//
#include <hip/hip_runtime.h>
#include <hip/hip_bf16.h>

// ---------------------------------------------------------------------------
// Problem constants (from reference)
// ---------------------------------------------------------------------------
#define N_NODES  25000
#define N_EDGES  400000
#define N_GRAPHS 64
#define F_IN     128
#define HID      512
// M padded to a multiple of 16 so WMMA fragment loads need no predication.
#define M_PAD    (((N_NODES + 15) / 16) * 16)   // 25008

// ---------------------------------------------------------------------------
// bf16 helpers (bit-level, round-to-nearest-even)
// ---------------------------------------------------------------------------
__device__ __forceinline__ unsigned short f32_to_bf16(float f) {
  unsigned int u = __float_as_uint(f);
  unsigned int rounding = 0x7fffu + ((u >> 16) & 1u);
  return (unsigned short)((u + rounding) >> 16);
}
__device__ __forceinline__ float bf16_to_f32(unsigned short h) {
  return __uint_as_float(((unsigned int)h) << 16);
}

// ---------------------------------------------------------------------------
// WMMA vector types (CDNA5 / gfx1250, wave32)
// ---------------------------------------------------------------------------
typedef __attribute__((ext_vector_type(16))) __bf16 v16bf;
typedef __attribute__((ext_vector_type(8)))  __bf16 v8bf;
typedef __attribute__((ext_vector_type(8)))  float  v8f;

// Load one 16x32 bf16 A/B fragment from a row-major [rows, K] bf16 matrix.
// ISA layout (16-bit A-matrix 16x32): lane holds row = lane%16,
// elements 0..7  = K (k0 + kb + 0..7),  elements 8..15 = K (k0 + kb + 16..23),
// where kb = 8 * (lane/16).  B uses the identical layout with row == column n
// of the (pre-transposed) weight matrix.  Rows are padded, so no predication.
__device__ __forceinline__ v16bf load_frag16x32(const unsigned short* base,
                                                int row, int K, int k0,
                                                int lane) {
  const __bf16* p = reinterpret_cast<const __bf16*>(base)
                  + (size_t)row * K + k0 + ((lane >> 4) << 3);
  v8bf lo = *reinterpret_cast<const v8bf*>(p);        // K = kb .. kb+7
  v8bf hi = *reinterpret_cast<const v8bf*>(p + 16);   // K = kb+16 .. kb+23
  v16bf f;
#pragma unroll
  for (int j = 0; j < 8; ++j) { f[j] = lo[j]; f[j + 8] = hi[j]; }
  return f;
}

// ---------------------------------------------------------------------------
// Fused dual-product GEMM:  D = act( Aa@Wa^T' + Ab@Wb^T' + bias )
//   Aa: [M_PAD,Ka] bf16 row-major   Wta: [N,Ka] bf16 (pre-transposed weight)
//   Ab: [M_PAD,Kb] bf16 row-major   Wtb: [N,Kb] bf16
// One wave -> one 16x16 output tile; 8 waves/block cover 128 output columns.
// All fragment loads are unconditional (M padded); only the store is guarded.
// ---------------------------------------------------------------------------
__global__ void gemm_dual_bf16_wmma(const unsigned short* __restrict__ Aa,
                                    const unsigned short* __restrict__ Wta, int Ka,
                                    const unsigned short* __restrict__ Ab,
                                    const unsigned short* __restrict__ Wtb, int Kb,
                                    const float* __restrict__ bias,
                                    float* __restrict__ outF,
                                    unsigned short* __restrict__ outB,
                                    int M, int N, int applyRelu) {
  const int lane  = threadIdx.x & 31;
  const int wave  = threadIdx.x >> 5;
  const int mBase = blockIdx.x * 16;
  const int nBase = (blockIdx.y * 8 + wave) * 16;
  if (nBase >= N) return;                       // wave-uniform exit

  const int mRow = mBase + (lane & 15);
  const int nRow = nBase + (lane & 15);         // column of W == row of Wt

  v8f c = {};
  for (int k0 = 0; k0 < Ka; k0 += 32) {
    v16bf a = load_frag16x32(Aa,  mRow, Ka, k0, lane);
    v16bf b = load_frag16x32(Wta, nRow, Ka, k0, lane);
    c = __builtin_amdgcn_wmma_f32_16x16x32_bf16(false, a, false, b,
                                                (short)0, c, false, false);
  }
  for (int k0 = 0; k0 < Kb; k0 += 32) {
    v16bf a = load_frag16x32(Ab,  mRow, Kb, k0, lane);
    v16bf b = load_frag16x32(Wtb, nRow, Kb, k0, lane);
    c = __builtin_amdgcn_wmma_f32_16x16x32_bf16(false, a, false, b,
                                                (short)0, c, false, false);
  }

  // C/D layout: VGPR r, lanes 0-15 -> (M=r, N=lane); lanes 16-31 -> (M=8+r).
  const int   nCol = nBase + (lane & 15);
  const float bv   = bias ? bias[nCol] : 0.0f;
  const int   mTop = mBase + ((lane >> 4) << 3);
#pragma unroll
  for (int r = 0; r < 8; ++r) {
    int m = mTop + r;
    if (m < M) {
      float v = c[r] + bv;
      if (applyRelu) v = fmaxf(v, 0.0f);
      if (outF) outF[(size_t)m * N + nCol] = v;
      if (outB) outB[(size_t)m * N + nCol] = f32_to_bf16(v);
    }
  }
}

// ---------------------------------------------------------------------------
// Conversion kernels
// ---------------------------------------------------------------------------
__global__ void cvt_f32_bf16(const float* __restrict__ src,
                             unsigned short* __restrict__ dst, int n) {
  int stride = gridDim.x * blockDim.x;
  for (int i = blockIdx.x * blockDim.x + threadIdx.x; i < n; i += stride)
    dst[i] = f32_to_bf16(src[i]);
}

// Wt[n*K + k] = bf16(W[k*N + n])   (transpose + convert; weights are small)
__global__ void cvtT_f32_bf16(const float* __restrict__ W,
                              unsigned short* __restrict__ Wt, int K, int N) {
  int total  = K * N;
  int stride = gridDim.x * blockDim.x;
  for (int i = blockIdx.x * blockDim.x + threadIdx.x; i < total; i += stride) {
    int k = i / N, n = i - k * N;
    Wt[(size_t)n * K + k] = f32_to_bf16(W[i]);
  }
}

// ---------------------------------------------------------------------------
// Edge gather-scale-scatter:  agg[dst] += w_e * X[src]
// FPT threads per edge, 4 features per thread (F = FPT*4).
// ---------------------------------------------------------------------------
template <int FPT>
__global__ void scatter_edges_f32(const float* __restrict__ X,
                                  const int* __restrict__ srcIdx,
                                  const int* __restrict__ dstIdx,
                                  const float* __restrict__ w,
                                  float* __restrict__ agg, int E) {
  long long tid = (long long)blockIdx.x * blockDim.x + threadIdx.x;
  if (tid >= (long long)E * FPT) return;
  int e  = (int)(tid / FPT);
  int f0 = ((int)(tid % FPT)) << 2;
  int s = srcIdx[e], d = dstIdx[e];
  float we = w[e];
  const float4 xv = *reinterpret_cast<const float4*>(X + (size_t)s * (FPT * 4) + f0);
  float* dp = agg + (size_t)d * (FPT * 4) + f0;
  atomicAdd(dp + 0, xv.x * we);
  atomicAdd(dp + 1, xv.y * we);
  atomicAdd(dp + 2, xv.z * we);
  atomicAdd(dp + 3, xv.w * we);
}

template <int FPT>
__global__ void scatter_edges_bf16(const unsigned short* __restrict__ X,
                                   const int* __restrict__ srcIdx,
                                   const int* __restrict__ dstIdx,
                                   const float* __restrict__ w,
                                   float* __restrict__ agg, int E) {
  long long tid = (long long)blockIdx.x * blockDim.x + threadIdx.x;
  if (tid >= (long long)E * FPT) return;
  int e  = (int)(tid / FPT);
  int f0 = ((int)(tid % FPT)) << 2;
  int s = srcIdx[e], d = dstIdx[e];
  float we = w[e];
  const ushort4 xv = *reinterpret_cast<const ushort4*>(X + (size_t)s * (FPT * 4) + f0);
  float* dp = agg + (size_t)d * (FPT * 4) + f0;
  atomicAdd(dp + 0, bf16_to_f32(xv.x) * we);
  atomicAdd(dp + 1, bf16_to_f32(xv.y) * we);
  atomicAdd(dp + 2, bf16_to_f32(xv.z) * we);
  atomicAdd(dp + 3, bf16_to_f32(xv.w) * we);
}

// ---------------------------------------------------------------------------
// Global mean pool (sum + count; division done in MLP stage)
// ---------------------------------------------------------------------------
__global__ void pool_sum(const float* __restrict__ h,
                         const int* __restrict__ batch,
                         float* __restrict__ sums, int Nn) {
  const int FPT = HID / 4;
  long long tid = (long long)blockIdx.x * blockDim.x + threadIdx.x;
  if (tid >= (long long)Nn * FPT) return;
  int node = (int)(tid / FPT);
  int f0   = ((int)(tid % FPT)) << 2;
  int g = batch[node];
  const float4 v = *reinterpret_cast<const float4*>(h + (size_t)node * HID + f0);
  float* sp = sums + (size_t)g * HID + f0;
  atomicAdd(sp + 0, v.x);
  atomicAdd(sp + 1, v.y);
  atomicAdd(sp + 2, v.z);
  atomicAdd(sp + 3, v.w);
}

__global__ void pool_count(const int* __restrict__ batch,
                           float* __restrict__ counts, int Nn) {
  int i = blockIdx.x * blockDim.x + threadIdx.x;
  if (i < Nn) atomicAdd(&counts[batch[i]], 1.0f);
}

// ---------------------------------------------------------------------------
// Tiny MLP head (per-graph, f32).  counts != nullptr -> divide input by count.
// ---------------------------------------------------------------------------
__global__ void mlp_layer(const float* __restrict__ in,
                          const float* __restrict__ Wl,
                          const float* __restrict__ bl,
                          const float* __restrict__ counts,
                          float* __restrict__ out,
                          int G, int Kin, int Nout, int applyRelu) {
  int tid = blockIdx.x * blockDim.x + threadIdx.x;
  if (tid >= G * Nout) return;
  int g = tid / Nout, j = tid - g * Nout;
  float inv = 1.0f;
  if (counts) inv = 1.0f / fmaxf(counts[g], 1.0f);
  float acc = bl[j];
  for (int k = 0; k < Kin; ++k)
    acc = fmaf(in[(size_t)g * Kin + k] * inv, Wl[(size_t)k * Nout + j], acc);
  if (applyRelu) acc = fmaxf(acc, 0.0f);
  out[(size_t)g * Nout + j] = acc;
}

// ---------------------------------------------------------------------------
// Host-side launcher
// ---------------------------------------------------------------------------
static inline size_t alignUp(size_t v) { return (v + 255) & ~(size_t)255; }

extern "C" void kernel_launch(void* const* d_in, const int* in_sizes, int n_in,
                              void* d_out, int out_size, void* d_ws, size_t ws_size,
                              hipStream_t stream) {
  (void)in_sizes; (void)n_in; (void)out_size; (void)ws_size;

  // Inputs (setup_inputs dict order)
  const float* x         = (const float*)d_in[0];     // [N_NODES, F_IN]
  const int*   edge_idx  = (const int*)d_in[1];       // [2, N_EDGES]
  const float* edge_attr = (const float*)d_in[2];     // [N_EDGES]
  const int*   batch_ids = (const int*)d_in[3];       // [N_NODES]
  const float* W1_rel    = (const float*)d_in[4];     // [F_IN, HID]
  const float* b1        = (const float*)d_in[5];     // [HID]
  const float* W1_root   = (const float*)d_in[6];     // [F_IN, HID]
  const float* W2_rel    = (const float*)d_in[7];     // [HID, HID]
  const float* b2        = (const float*)d_in[8];     // [HID]
  const float* W2_root   = (const float*)d_in[9];     // [HID, HID]
  const float* Wl1       = (const float*)d_in[10];    // [HID, 64]
  const float* bl1       = (const float*)d_in[11];    // [64]
  const float* Wl2       = (const float*)d_in[12];    // [64, 16]
  const float* bl2       = (const float*)d_in[13];    // [16]
  const float* Wl3       = (const float*)d_in[14];    // [16, 1]
  const float* bl3       = (const float*)d_in[15];    // [1]

  const int* srcIdx = edge_idx;             // edge_index[0, :]
  const int* dstIdx = edge_idx + N_EDGES;   // edge_index[1, :]

  // Workspace bump allocator (activation buffers padded to M_PAD rows)
  char* ws = (char*)d_ws;
  size_t off = 0;
  auto carve = [&](size_t bytes) -> void* {
    void* p = ws + off; off += alignUp(bytes); return p;
  };
  unsigned short* xb      = (unsigned short*)carve((size_t)M_PAD * F_IN * 2);
  unsigned short* wt1rel  = (unsigned short*)carve((size_t)HID * F_IN * 2);
  unsigned short* wt1root = (unsigned short*)carve((size_t)HID * F_IN * 2);
  unsigned short* wt2rel  = (unsigned short*)carve((size_t)HID * HID * 2);
  unsigned short* wt2root = (unsigned short*)carve((size_t)HID * HID * 2);
  float*          agg1    = (float*)carve((size_t)M_PAD * F_IN * 4);
  unsigned short* agg1b   = (unsigned short*)carve((size_t)M_PAD * F_IN * 2);
  unsigned short* h1b     = (unsigned short*)carve((size_t)M_PAD * HID * 2);
  float*          agg2    = (float*)carve((size_t)M_PAD * HID * 4);
  unsigned short* agg2b   = (unsigned short*)carve((size_t)M_PAD * HID * 2);
  float*          h2      = (float*)carve((size_t)N_NODES * HID * 4);
  float*          sums    = (float*)carve((size_t)N_GRAPHS * HID * 4);
  float*          counts  = (float*)carve((size_t)N_GRAPHS * 4);
  float*          g1      = (float*)carve((size_t)N_GRAPHS * 64 * 4);
  float*          g2      = (float*)carve((size_t)N_GRAPHS * 16 * 4);

  // Zero the atomic accumulators (full padded range) and the tiny pad tails
  // of bf16 activation buffers so WMMA never reads poisoned bits.
  hipMemsetAsync(agg1,   0, (size_t)M_PAD * F_IN * 4, stream);
  hipMemsetAsync(agg2,   0, (size_t)M_PAD * HID * 4, stream);
  hipMemsetAsync(sums,   0, (size_t)N_GRAPHS * HID * 4, stream);
  hipMemsetAsync(counts, 0, (size_t)N_GRAPHS * 4, stream);
  hipMemsetAsync(xb  + (size_t)N_NODES * F_IN, 0,
                 (size_t)(M_PAD - N_NODES) * F_IN * 2, stream);
  hipMemsetAsync(h1b + (size_t)N_NODES * HID, 0,
                 (size_t)(M_PAD - N_NODES) * HID * 2, stream);

  const int BLK = 256;
  auto blocksFor = [&](long long n) { return (unsigned)((n + BLK - 1) / BLK); };

  // --- precision conversions (x and weights -> bf16; weights transposed) ---
  cvt_f32_bf16<<<blocksFor((long long)N_NODES * F_IN), BLK, 0, stream>>>(
      x, xb, N_NODES * F_IN);
  cvtT_f32_bf16<<<blocksFor((long long)F_IN * HID), BLK, 0, stream>>>(
      W1_rel, wt1rel, F_IN, HID);
  cvtT_f32_bf16<<<blocksFor((long long)F_IN * HID), BLK, 0, stream>>>(
      W1_root, wt1root, F_IN, HID);
  cvtT_f32_bf16<<<blocksFor((long long)HID * HID), BLK, 0, stream>>>(
      W2_rel, wt2rel, HID, HID);
  cvtT_f32_bf16<<<blocksFor((long long)HID * HID), BLK, 0, stream>>>(
      W2_root, wt2root, HID, HID);

  // --- layer 1: scatter-add messages, then fused dual GEMM + bias + ReLU ---
  scatter_edges_f32<F_IN / 4><<<blocksFor((long long)N_EDGES * (F_IN / 4)),
                                BLK, 0, stream>>>(x, srcIdx, dstIdx, edge_attr,
                                                  agg1, N_EDGES);
  cvt_f32_bf16<<<blocksFor((long long)M_PAD * F_IN), BLK, 0, stream>>>(
      agg1, agg1b, M_PAD * F_IN);   // pad rows are zeros from memset

  {
    dim3 grid(M_PAD / 16, HID / 128);   // 8 waves * 16 cols = 128
    gemm_dual_bf16_wmma<<<grid, BLK, 0, stream>>>(
        agg1b, wt1rel, F_IN, xb, wt1root, F_IN,
        b1, nullptr, h1b, N_NODES, HID, /*relu=*/1);
  }

  // --- layer 2 ---
  scatter_edges_bf16<HID / 4><<<blocksFor((long long)N_EDGES * (HID / 4)),
                                BLK, 0, stream>>>(h1b, srcIdx, dstIdx, edge_attr,
                                                  agg2, N_EDGES);
  cvt_f32_bf16<<<blocksFor((long long)M_PAD * HID), BLK, 0, stream>>>(
      agg2, agg2b, M_PAD * HID);    // pad rows are zeros from memset

  {
    dim3 grid(M_PAD / 16, HID / 128);
    gemm_dual_bf16_wmma<<<grid, BLK, 0, stream>>>(
        agg2b, wt2rel, HID, h1b, wt2root, HID,
        b2, h2, nullptr, N_NODES, HID, /*relu=*/1);
  }

  // --- global mean pool ---
  pool_sum<<<blocksFor((long long)N_NODES * (HID / 4)), BLK, 0, stream>>>(
      h2, batch_ids, sums, N_NODES);
  pool_count<<<blocksFor(N_NODES), BLK, 0, stream>>>(batch_ids, counts, N_NODES);

  // --- MLP head: 512 -> 64 -> 16 -> 1 ---
  mlp_layer<<<blocksFor(N_GRAPHS * 64), BLK, 0, stream>>>(
      sums, Wl1, bl1, counts, g1, N_GRAPHS, HID, 64, /*relu=*/1);
  mlp_layer<<<blocksFor(N_GRAPHS * 16), BLK, 0, stream>>>(
      g1, Wl2, bl2, nullptr, g2, N_GRAPHS, 64, 16, /*relu=*/1);
  mlp_layer<<<blocksFor(N_GRAPHS), BLK, 0, stream>>>(
      g2, Wl3, bl3, nullptr, (float*)d_out, N_GRAPHS, 16, 1, /*relu=*/0);
}
